// GCN3_6751688589933
// MI455X (gfx1250) — matrix operand order repaired and checked by reading, not compiled, not run on hardware
//
#include <hip/hip_runtime.h>

// ---------------- problem constants (from reference setup_inputs) ----------
#define N_NODES 30000
#define N_EDGES 480000
#define ETOT    (N_EDGES + N_NODES)   // self loops appended
#define FIN     16
#define HH      8
#define CC      64
#define HCDIM   512                   // H*C
#define NG      64                    // graphs
#define NEG_SLOPE 0.2f

typedef __attribute__((ext_vector_type(16))) _Float16 v16h;
typedef __attribute__((ext_vector_type(8)))  _Float16 v8h;
typedef __attribute__((ext_vector_type(8)))  float    v8f;

union FragA { v16h v; v8h h[2]; };

// ------------- helpers ----------------
__device__ inline unsigned fenc(float f) {
    unsigned b = __float_as_uint(f);
    return (b & 0x80000000u) ? ~b : (b | 0x80000000u);
}
__device__ inline float fdec(unsigned u) {
    return (u & 0x80000000u) ? __uint_as_float(u & 0x7FFFFFFFu)
                             : __uint_as_float(~u);
}

// ------------- BatchNorm statistics: sum / sumsq per feature --------------
__global__ void bn_stats_kernel(const float* __restrict__ x, float* __restrict__ stats) {
    __shared__ float ls[32];
    int tid = threadIdx.x;
    if (tid < 32) ls[tid] = 0.f;
    __syncthreads();
    for (int t = blockIdx.x * blockDim.x + tid; t < N_NODES * FIN;
         t += gridDim.x * blockDim.x) {
        int f = t & 15;
        float v = x[t];
        atomicAdd(&ls[f], v);
        atomicAdd(&ls[16 + f], v * v);
    }
    __syncthreads();
    if (tid < 32) atomicAdd(&stats[tid], ls[tid]);
}

// ------------- BatchNorm apply -> f16, K padded to 32 ---------------------
__global__ void bn_apply_kernel(const float* __restrict__ x, const float* __restrict__ stats,
                                const float* __restrict__ gamma, const float* __restrict__ beta,
                                _Float16* __restrict__ xn16) {
    int t = blockIdx.x * blockDim.x + threadIdx.x;
    if (t >= N_NODES * FIN) return;
    int f = t & 15, n = t >> 4;
    const float invN = 1.0f / (float)N_NODES;
    float mu  = stats[f] * invN;
    float var = stats[16 + f] * invN - mu * mu;
    float v = (x[t] - mu) * rsqrtf(var + 1e-5f) * gamma[f] + beta[f];
    xn16[(size_t)n * 32 + f] = (_Float16)v;   // cols 16..31 pre-zeroed
}

// ------------- transpose + convert weight: W[K x 512] -> Wt[512 x Kp] f16 -
__global__ void wtrans_kernel(const float* __restrict__ W, _Float16* __restrict__ Wt,
                              int Kact, int Kp) {
    int t = blockIdx.x * blockDim.x + threadIdx.x;
    if (t >= HCDIM * Kact) return;
    int nOut = t / Kact, k = t - nOut * Kact;
    Wt[(size_t)nOut * Kp + k] = (_Float16)W[(size_t)k * HCDIM + nOut];
}

// ------------- WMMA GEMM: h[M x 512] = A[M x Kp](f16) * Wt^T --------------
// one wave -> one 16x16 tile of D ; 4 waves per block (wave32)
__global__ void wmma_gemm_kernel(const _Float16* __restrict__ A,
                                 const _Float16* __restrict__ Bt,
                                 float* __restrict__ Cm, int lda) {
    int wave  = threadIdx.x >> 5;
    int lane  = threadIdx.x & 31;
    int ntile = blockIdx.x * 4 + wave;      // 32 N-tiles
    int mtile = blockIdx.y;                 // 1875 M-tiles
    int hid   = lane >> 4;                  // 0: lanes 0-15, 1: lanes 16-31
    int r     = lane & 15;

    // A fragment rows: lane holds row (mtile*16+r); K split contiguous, k = 16*hid + s
    const _Float16* arow = A  + (size_t)(mtile * 16 + r) * lda + hid * 16;
    // B fragment: lane holds column n = ntile*16+r; 16 contiguous K values
    const _Float16* brow = Bt + (size_t)(ntile * 16 + r) * lda + hid * 16;

    v8f acc = {};
    for (int kk = 0; kk < lda; kk += 32) {
        __builtin_prefetch(arow + kk + 64, 0, 1);   // global_prefetch_b8
        FragA a;
        a.h[0] = *(const v8h*)(arow + kk);          // K kk+{0..7}  (or +16..23)
        a.h[1] = *(const v8h*)(arow + kk + 8);      // K kk+{8..15} (or 24..31)
        v16h b = *(const v16h*)(brow + kk);         // K kk+{0..15} (or 16..31)
        acc = __builtin_amdgcn_wmma_f32_16x16x32_f16(
            /*neg_a=*/false, a.v, /*neg_b=*/false, b,
            /*c_mod=*/(short)0, acc, /*reuse_a=*/false, /*reuse_b=*/false);
    }
    // D layout: VGPR i -> row (mtile*16 + hid*8 + i), col ntile*16 + r
    float* cp = Cm + (size_t)(mtile * 16 + hid * 8) * HCDIM + (ntile * 16 + r);
#pragma unroll
    for (int i = 0; i < 8; ++i) cp[(size_t)i * HCDIM] = acc[i];
}

// ------------- per-node attention logits: s[n,h], d[n,h] ------------------
__global__ void logits_kernel(const float* __restrict__ hfeat,
                              const float* __restrict__ asrc, const float* __restrict__ adst,
                              float* __restrict__ slog, float* __restrict__ dlog) {
    int t = blockIdx.x * blockDim.x + threadIdx.x;
    if (t >= N_NODES * HH) return;
    int n = t >> 3, hh = t & 7;
    const float* hp = hfeat + (size_t)n * HCDIM + hh * CC;
    const float* as = asrc + hh * CC;
    const float* ad = adst + hh * CC;
    float s = 0.f, d = 0.f;
#pragma unroll 8
    for (int c = 0; c < CC; ++c) { s += hp[c] * as[c]; d += hp[c] * ad[c]; }
    slog[t] = s; dlog[t] = d;
}

// ------------- edge pass 1: segment max of leaky-relu logits --------------
__global__ void edge_max_kernel(const int* __restrict__ ei,
                                const float* __restrict__ slog, const float* __restrict__ dlog,
                                unsigned* __restrict__ emax) {
    int t = blockIdx.x * blockDim.x + threadIdx.x;
    if (t >= ETOT * HH) return;
    int eid = t >> 3, hh = t & 7;
    int s = (eid < N_EDGES) ? ei[eid]            : (eid - N_EDGES);
    int d = (eid < N_EDGES) ? ei[N_EDGES + eid]  : (eid - N_EDGES);
    float e = slog[s * HH + hh] + dlog[d * HH + hh];
    e = (e > 0.f) ? e : NEG_SLOPE * e;
    atomicMax(&emax[d * HH + hh], fenc(e));      // global_atomic_max_u32
}

// ------------- edge pass 2: exp + segment sum -----------------------------
__global__ void edge_exp_kernel(const int* __restrict__ ei,
                                const float* __restrict__ slog, const float* __restrict__ dlog,
                                const unsigned* __restrict__ emax,
                                float* __restrict__ denom, float* __restrict__ exb) {
    int t = blockIdx.x * blockDim.x + threadIdx.x;
    if (t >= ETOT * HH) return;
    int eid = t >> 3, hh = t & 7;
    int s = (eid < N_EDGES) ? ei[eid]            : (eid - N_EDGES);
    int d = (eid < N_EDGES) ? ei[N_EDGES + eid]  : (eid - N_EDGES);
    float e = slog[s * HH + hh] + dlog[d * HH + hh];
    e = (e > 0.f) ? e : NEG_SLOPE * e;
    float ex = __expf(e - fdec(emax[d * HH + hh]));
    atomicAdd(&denom[d * HH + hh], ex);          // global_atomic_add_f32
    exb[t] = ex;
}

// ------------- weighted message aggregation: one block per edge -----------
__global__ void agg_kernel(const int* __restrict__ ei, const float* __restrict__ hfeat,
                           const float* __restrict__ exb, const float* __restrict__ denom,
                           float* __restrict__ agg) {
    int eid = blockIdx.x;
    __shared__ float s_alpha[HH];
    int s = (eid < N_EDGES) ? ei[eid]            : (eid - N_EDGES);
    int d = (eid < N_EDGES) ? ei[N_EDGES + eid]  : (eid - N_EDGES);
    int tid = threadIdx.x;
    if (tid < HH)
        s_alpha[tid] = exb[(size_t)eid * HH + tid] / (denom[d * HH + tid] + 1e-16f);
    __syncthreads();
    int idx = tid * 2;                              // 256 threads * 2 channels
    float2 hv = *(const float2*)(hfeat + (size_t)s * HCDIM + idx);
    float al = s_alpha[idx >> 6];                   // pair never crosses a head
    atomicAdd(agg + (size_t)d * HCDIM + idx,     hv.x * al);
    atomicAdd(agg + (size_t)d * HCDIM + idx + 1, hv.y * al);
}

// ------------- head mean + bias (+relu); emit f16 (next layer) & f32 ------
__global__ void head_mean_kernel(const float* __restrict__ agg, const float* __restrict__ bias,
                                 _Float16* __restrict__ x16, float* __restrict__ x32,
                                 int relu) {
    int t = blockIdx.x * blockDim.x + threadIdx.x;
    if (t >= N_NODES * CC) return;
    int n = t >> 6, c = t & 63;
    const float* ap = agg + (size_t)n * HCDIM + c;
    float v = 0.f;
#pragma unroll
    for (int h = 0; h < HH; ++h) v += ap[h * CC];
    v = v * 0.125f + bias[c];
    if (relu) v = fmaxf(v, 0.f);
    x16[t] = (_Float16)v;
    x32[t] = v;
}

// ------------- global mean pool (segment sum + counts) --------------------
__global__ void pool_kernel(const float* __restrict__ x32, const int* __restrict__ batch,
                            float* __restrict__ pooled, float* __restrict__ cnt) {
    int t = blockIdx.x * blockDim.x + threadIdx.x;
    if (t >= N_NODES * CC) return;
    int n = t >> 6, c = t & 63;
    int b = batch[n];
    atomicAdd(&pooled[b * CC + c], x32[t]);
    if (c == 0) atomicAdd(&cnt[b], 1.0f);
}

// ------------- final linear 64 -> 2 ---------------------------------------
__global__ void final_kernel(const float* __restrict__ pooled, const float* __restrict__ cnt,
                             const float* __restrict__ linW, const float* __restrict__ linb,
                             float* __restrict__ out) {
    int t = threadIdx.x;
    if (t >= NG * 2) return;
    int g = t >> 1, j = t & 1;
    float inv = 1.0f / fmaxf(cnt[g], 1.0f);
    float acc = linb[j];
    for (int c = 0; c < CC; ++c)
        acc += pooled[g * CC + c] * inv * linW[c * 2 + j];
    out[t] = acc;
}

// =================== workspace layout (bytes, 256-aligned) ================
#define OFF_STATS  ((size_t)0)            //   128
#define OFF_XN16   ((size_t)256)          // 30000*32*2 = 1,920,000
#define OFF_WT16   ((size_t)1920256)      // 512*64*2   =    65,536
#define OFF_H      ((size_t)1985792)      // 30000*512*4= 61,440,000
#define OFF_SLOG   ((size_t)63425792)     // 30000*8*4  =   960,000
#define OFF_DLOG   ((size_t)64385792)
#define OFF_EMAX   ((size_t)65345792)
#define OFF_DENOM  ((size_t)66305792)
#define OFF_EX     ((size_t)67265792)     // 510000*8*4 = 16,320,000
#define OFF_AGG    ((size_t)83585792)     // 61,440,000
#define OFF_X16    ((size_t)145025792)    // 30000*64*2 =  3,840,000
#define OFF_X3     ((size_t)148865792)    // 30000*64*4 =  7,680,000
#define OFF_POOL   ((size_t)156545792)    // 64*64*4
#define OFF_CNT    ((size_t)156562176)    // 64*4 (pad 256)

extern "C" void kernel_launch(void* const* d_in, const int* in_sizes, int n_in,
                              void* d_out, int out_size, void* d_ws, size_t ws_size,
                              hipStream_t stream) {
    const float* x     = (const float*)d_in[0];
    const int*   ei    = (const int*)  d_in[1];
    const int*   batch = (const int*)  d_in[3];
    const float* gamma = (const float*)d_in[4];
    const float* beta  = (const float*)d_in[5];
    const float* W[3]    = { (const float*)d_in[6],  (const float*)d_in[10], (const float*)d_in[14] };
    const float* Asrc[3] = { (const float*)d_in[7],  (const float*)d_in[11], (const float*)d_in[15] };
    const float* Adst[3] = { (const float*)d_in[8],  (const float*)d_in[12], (const float*)d_in[16] };
    const float* Bias[3] = { (const float*)d_in[9],  (const float*)d_in[13], (const float*)d_in[17] };
    const float* linW  = (const float*)d_in[18];
    const float* linb  = (const float*)d_in[19];

    char* ws = (char*)d_ws;
    float*     stats = (float*)(ws + OFF_STATS);
    _Float16*  xn16  = (_Float16*)(ws + OFF_XN16);
    _Float16*  wt16  = (_Float16*)(ws + OFF_WT16);
    float*     hbuf  = (float*)(ws + OFF_H);
    float*     slog  = (float*)(ws + OFF_SLOG);
    float*     dlog  = (float*)(ws + OFF_DLOG);
    unsigned*  emax  = (unsigned*)(ws + OFF_EMAX);
    float*     denom = (float*)(ws + OFF_DENOM);
    float*     exb   = (float*)(ws + OFF_EX);
    float*     agg   = (float*)(ws + OFF_AGG);
    _Float16*  x16   = (_Float16*)(ws + OFF_X16);
    float*     x3    = (float*)(ws + OFF_X3);
    float*     pooled= (float*)(ws + OFF_POOL);
    float*     cnt   = (float*)(ws + OFF_CNT);

    const int TB = 256;

    // ---- BatchNorm ----
    hipMemsetAsync(stats, 0, 128, stream);
    hipMemsetAsync(xn16, 0, (size_t)N_NODES * 32 * 2, stream);   // padded K zeros
    bn_stats_kernel<<<480, TB, 0, stream>>>(x, stats);
    bn_apply_kernel<<<(N_NODES * FIN + TB - 1) / TB, TB, 0, stream>>>(x, stats, gamma, beta, xn16);

    // ---- 3 GAT layers ----
    for (int L = 0; L < 3; ++L) {
        const int Kact = (L == 0) ? FIN : CC;
        const int Kp   = (L == 0) ? 32  : CC;           // pad K=16 -> 32 for WMMA
        const _Float16* A16 = (L == 0) ? xn16 : x16;

        // weight transpose/convert (zero-padded)
        hipMemsetAsync(wt16, 0, (size_t)HCDIM * Kp * 2, stream);
        wtrans_kernel<<<(HCDIM * Kact + TB - 1) / TB, TB, 0, stream>>>(W[L], wt16, Kact, Kp);

        // projection via v_wmma_f32_16x16x32_f16 ; grid: 8 x 1875, 4 waves/blk
        wmma_gemm_kernel<<<dim3(HCDIM / 16 / 4, N_NODES / 16), 128, 0, stream>>>(A16, wt16, hbuf, Kp);

        // attention
        logits_kernel<<<(N_NODES * HH + TB - 1) / TB, TB, 0, stream>>>(hbuf, Asrc[L], Adst[L], slog, dlog);
        hipMemsetAsync(emax, 0, (size_t)N_NODES * HH * 4, stream);
        hipMemsetAsync(denom, 0, (size_t)N_NODES * HH * 4, stream);
        edge_max_kernel<<<((size_t)ETOT * HH + TB - 1) / TB, TB, 0, stream>>>(ei, slog, dlog, emax);
        edge_exp_kernel<<<((size_t)ETOT * HH + TB - 1) / TB, TB, 0, stream>>>(ei, slog, dlog, emax, denom, exb);

        // aggregation
        hipMemsetAsync(agg, 0, (size_t)N_NODES * HCDIM * 4, stream);
        agg_kernel<<<ETOT, TB, 0, stream>>>(ei, hbuf, exb, denom, agg);

        // head mean + bias (+relu on layers 0,1)
        head_mean_kernel<<<(N_NODES * CC + TB - 1) / TB, TB, 0, stream>>>(agg, Bias[L], x16, x3, (L < 2) ? 1 : 0);
    }

    // ---- global mean pool + final linear ----
    hipMemsetAsync(pooled, 0, (size_t)NG * CC * 4 + 256, stream);   // pooled + cnt (contiguous)
    pool_kernel<<<(N_NODES * CC + TB - 1) / TB, TB, 0, stream>>>(x3, batch, pooled, cnt);
    final_kernel<<<1, 128, 0, stream>>>(pooled, cnt, linW, linb, (float*)d_out);
}
